// PoolerNeighborMap_77232101916960
// MI455X (gfx1250) — matrix operand order repaired and checked by reading, not compiled, not run on hardware
//
#include <hip/hip_runtime.h>
#include <hip/hip_bf16.h>
#include <stdint.h>

typedef __attribute__((ext_vector_type(16))) _Float16 v16h;
typedef __attribute__((ext_vector_type(8)))  float    v8f;

#define POOLEDN   10
#define SGRID     20          // POOLED * SR sample grid per axis
#define NSAMP     400         // 20*20 real samples
#define NSAMP_PAD 448         // padded with zero-weight entries -> branchless inner loop
#define NBOX      512
#define NCH       256
#define CHUNK     16          // channels per pass (WMMA N)
#define AREA_CAP  1024        // max patch elems per channel per LDS buffer
#define NEXPAND   1.5f

__shared__ int    sYLo[SGRID], sYHi[SGRID], sXLo[SGRID], sXHi[SGRID];
__shared__ float  sLy[SGRID],  sLx[SGRID],  sVy[SGRID],  sVx[SGRID];
__shared__ int4   descO[NSAMP_PAD];             // tap offsets (b128 aligned)
__shared__ float4 descW[NSAMP_PAD];             // premultiplied bilinear*valid weights
__shared__ float  patch[2][CHUNK * (AREA_CAP + 1)];

// B fragment from LDS patch: 16 consecutive samples, straight-line ds loads.
__device__ __forceinline__ v16h buildB_lds(int sBase, int cbuf, int chOff) {
    v16h B;
#pragma unroll
    for (int e = 0; e < 16; ++e) {
        const int s = sBase + e;
        const int4   o = descO[s];
        const float4 g = descW[s];
        const float f00 = patch[cbuf][chOff + o.x];
        const float f01 = patch[cbuf][chOff + o.y];
        const float f10 = patch[cbuf][chOff + o.z];
        const float f11 = patch[cbuf][chOff + o.w];
        B[e] = (_Float16)(g.x * f00 + g.y * f01 + g.z * f10 + g.w * f11);
    }
    return B;
}

// B fragment from global (oversized-patch fallback; features are L2-resident).
__device__ __forceinline__ v16h buildB_glb(int sBase, const float* __restrict__ gchan) {
    v16h B;
#pragma unroll
    for (int e = 0; e < 16; ++e) {
        const int s = sBase + e;
        const int4   o = descO[s];
        const float4 g = descW[s];
        const float f00 = gchan[o.x];
        const float f01 = gchan[o.y];
        const float f10 = gchan[o.z];
        const float f11 = gchan[o.w];
        B[e] = (_Float16)(g.x * f00 + g.y * f01 + g.z * f10 + g.w * f11);
    }
    return B;
}

// One workgroup per box. 256 threads = 8 wave32.
// Pipeline: async-stage chunk c+1 into buf[(c+1)&1] while waves 0..6 compute
// WMMA pooling tiles for chunk c from buf[c&1].
__global__ __launch_bounds__(256)
void pooler_neighbor_kernel(const float* __restrict__ x0,
                            const float* __restrict__ x1,
                            const float* __restrict__ x2,
                            const float* __restrict__ x3,
                            const float* __restrict__ boxes,
                            const int*   __restrict__ batch_ids,
                            float*       __restrict__ out)
{
    const int n    = blockIdx.x;
    const int tid  = threadIdx.x;
    const int lane = tid & 31;
    const int wave = tid >> 5;

    // ---- per-box scalars (uniform across block) ----
    const float bx1 = boxes[n * 4 + 0], by1 = boxes[n * 4 + 1];
    const float bx2 = boxes[n * 4 + 2], by2 = boxes[n * 4 + 3];
    const float bw = bx2 - bx1 + 1.0f, bh = by2 - by1 + 1.0f;
    const float sdiag = sqrtf(bw * bh);
    const int   lv = (sdiag < 112.0f) ? 0 : (sdiag < 224.0f) ? 1 : (sdiag < 448.0f) ? 2 : 3;

    const float scales[4] = {0.25f, 0.125f, 0.0625f, 0.03125f};
    const int   Hs[4]     = {200, 100, 50, 25};
    const int   Ws[4]     = {304, 152, 76, 38};
    const float masklv[4] = {0.25f, 0.5f, 1.0f, 2.0f};

    const float scale = scales[lv];
    const int   H = Hs[lv], W = Ws[lv];
    const float* feat = (lv == 0) ? x0 : (lv == 1) ? x1 : (lv == 2) ? x2 : x3;
    const int   b = batch_ids[n];

    const float cx = bx1 + 0.5f * bw, cy = by1 + 0.5f * bh;
    const float ew = bw * NEXPAND,    eh = bh * NEXPAND;
    const float rx1 = (cx - 0.5f * ew) * scale;
    const float ry1 = (cy - 0.5f * eh) * scale;
    const float rx2 = (cx + 0.5f * ew - 1.0f) * scale;
    const float ry2 = (cy + 0.5f * eh - 1.0f) * scale;
    const float rw = fmaxf(rx2 - rx1, 1.0f);
    const float rh = fmaxf(ry2 - ry1, 1.0f);

    if (tid == 0) out[(size_t)NBOX * NCH * POOLEDN * POOLEDN + n] = masklv[lv];

    // ---- per-axis sample prep ----
    if (tid < SGRID) {
        const int i = tid;
        const float c = rx1 + rw * (((float)i + 0.5f) / (float)SGRID);
        const float valid = (c > -1.0f && c < (float)W) ? 1.0f : 0.0f;
        const float cc = fmaxf(c, 0.0f);
        const float cf = floorf(cc);
        const int lo = (int)fminf(cf, (float)(W - 1));
        sXLo[i] = lo; sXHi[i] = min(lo + 1, W - 1);
        sLx[i]  = (cf >= (float)(W - 1)) ? 0.0f : (cc - cf);
        sVx[i]  = valid;
    } else if (tid >= 32 && tid < 32 + SGRID) {
        const int i = tid - 32;
        const float c = ry1 + rh * (((float)i + 0.5f) / (float)SGRID);
        const float valid = (c > -1.0f && c < (float)H) ? 1.0f : 0.0f;
        const float cc = fmaxf(c, 0.0f);
        const float cf = floorf(cc);
        const int lo = (int)fminf(cf, (float)(H - 1));
        sYLo[i] = lo; sYHi[i] = min(lo + 1, H - 1);
        sLy[i]  = (cf >= (float)(H - 1)) ? 0.0f : (cc - cf);
        sVy[i]  = valid;
    }
    __syncthreads();

    const int px0 = sXLo[0], py0 = sYLo[0];
    const int Wp  = sXHi[SGRID - 1] - px0 + 1;
    const int Hp  = sYHi[SGRID - 1] - py0 + 1;
    const int area = Wp * Hp;
    const bool useLds = (area <= AREA_CAP);
    const int chStride = area | 1;              // odd stride: conflict-free over channels

    const size_t HW = (size_t)H * (size_t)W;
    const float* featB = feat + (size_t)b * NCH * HW;

    // ---- build per-sample tap descriptors (chunk-invariant); pad with zeros ----
    {
        const int rowStride = useLds ? Wp : W;
        const int oy = useLds ? py0 : 0, ox = useLds ? px0 : 0;
        for (int s = tid; s < NSAMP_PAD; s += 256) {
            if (s < NSAMP) {
                const int pool = s >> 2, q = s & 3;
                const int ph = pool / 10;
                const int iy = ph * 2 + (q >> 1);
                const int ix = (pool - ph * 10) * 2 + (q & 1);
                const int r0 = (sYLo[iy] - oy) * rowStride;
                const int r1 = (sYHi[iy] - oy) * rowStride;
                const int cl = sXLo[ix] - ox, ch = sXHi[ix] - ox;
                descO[s] = make_int4(r0 + cl, r0 + ch, r1 + cl, r1 + ch);
                const float ly = sLy[iy], lx = sLx[ix];
                const float hy = 1.0f - ly, hx = 1.0f - lx;
                const float vv = sVy[iy] * sVx[ix];
                descW[s] = make_float4(hy * hx * vv, hy * lx * vv, ly * hx * vv, ly * lx * vv);
            } else {
                descO[s] = make_int4(0, 0, 0, 0);
                descW[s] = make_float4(0.f, 0.f, 0.f, 0.f);   // zero weight -> exact 0
            }
        }
    }

    // ---- async staging helper (div-free, row-contiguous) ----
    auto stage = [&](int c0, int buf) {
        for (int ch = wave; ch < CHUNK; ch += 8) {
            const float* g0 = featB + (size_t)(c0 + ch) * HW + (size_t)py0 * W + px0;
            const unsigned l0 = (unsigned)(uintptr_t)(const void*)&patch[buf][ch * chStride];
            for (int y = 0; y < Hp; ++y) {
                const float* gr = g0 + (size_t)y * W;
                const unsigned lr = l0 + (unsigned)(y * Wp) * 4u;
                for (int x = lane; x < Wp; x += 32) {
                    const unsigned la = lr + (unsigned)x * 4u;
                    const unsigned long long ga = (unsigned long long)(uintptr_t)(gr + x);
                    asm volatile("global_load_async_to_lds_b32 %0, %1, off"
                                 :: "v"(la), "v"(ga) : "memory");
                }
            }
        }
    };

    // ---- constant A fragments: pooling weights 0.25 (exact in f16) ----
    const int tile = wave;                        // waves 0..6 compute
    const int m0   = tile * 16;
    const int mrow = lane & 15;
    const int chn  = lane & 15;
    v16h A0, A1;
#pragma unroll
    for (int e = 0; e < 16; ++e) {
        const int Ka = e + ((lane >= 16) ? 16 : 0);
        const int s0 = 4 * m0 + Ka;
        A0[e] = ((s0 >> 2)        == (m0 + mrow)) ? (_Float16)0.25f : (_Float16)0.0f;
        A1[e] = (((s0 + 32) >> 2) == (m0 + mrow)) ? (_Float16)0.25f : (_Float16)0.0f;
    }

    // ---- pipeline prologue ----
    if (useLds) {
        stage(0, 0);
        asm volatile("s_wait_asynccnt 0x0" ::: "memory");
    }
    __syncthreads();

    for (int c = 0; c < NCH / CHUNK; ++c) {
        const int c0 = c * CHUNK;
        if (useLds && c + 1 < NCH / CHUNK)
            stage(c0 + CHUNK, (c + 1) & 1);       // overlap async DMA with compute

        if (tile < 7) {
            const int cbuf  = c & 1;
            const int chOff = chn * chStride;
            const int sb0   = 4 * m0 + ((lane >= 16) ? 16 : 0);
            v8f acc = {0.f, 0.f, 0.f, 0.f, 0.f, 0.f, 0.f, 0.f};

            // k-steps manually unrolled: fixed A operands, no cndmask selection.
            if (useLds) {
                const v16h B0 = buildB_lds(sb0, cbuf, chOff);
                acc = __builtin_amdgcn_wmma_f32_16x16x32_f16(
                          false, A0, false, B0, (short)0, acc, false, false);
                if (tile != 6) {                  // tile6 k-step1 is all zero-weight
                    const v16h B1 = buildB_lds(sb0 + 32, cbuf, chOff);
                    acc = __builtin_amdgcn_wmma_f32_16x16x32_f16(
                              false, A1, false, B1, (short)0, acc, false, false);
                }
            } else {
                const float* __restrict__ gchan = featB + (size_t)(c0 + chn) * HW;
                const v16h B0 = buildB_glb(sb0, gchan);
                acc = __builtin_amdgcn_wmma_f32_16x16x32_f16(
                          false, A0, false, B0, (short)0, acc, false, false);
                if (tile != 6) {
                    const v16h B1 = buildB_glb(sb0 + 32, gchan);
                    acc = __builtin_amdgcn_wmma_f32_16x16x32_f16(
                              false, A1, false, B1, (short)0, acc, false, false);
                }
            }

            // D: VGPR r -> pool m0+r (lanes 0-15) / m0+8+r (lanes 16-31), N = channel
#pragma unroll
            for (int r = 0; r < 8; ++r) {
                const int pool = m0 + r + ((lane >= 16) ? 8 : 0);
                if (pool < 100) {
                    const int ch = c0 + (lane & 15);
                    __builtin_nontemporal_store(acc[r],
                        &out[((size_t)n * NCH + ch) * 100 + pool]);
                }
            }
        }

        if (useLds) asm volatile("s_wait_asynccnt 0x0" ::: "memory");
        __syncthreads();
    }
}

extern "C" void kernel_launch(void* const* d_in, const int* in_sizes, int n_in,
                              void* d_out, int out_size, void* d_ws, size_t ws_size,
                              hipStream_t stream) {
    const float* x0    = (const float*)d_in[0];
    const float* x1    = (const float*)d_in[1];
    const float* x2    = (const float*)d_in[2];
    const float* x3    = (const float*)d_in[3];
    const float* boxes = (const float*)d_in[4];
    const int*   bids  = (const int*)d_in[5];
    float* out = (float*)d_out;
    hipLaunchKernelGGL(pooler_neighbor_kernel, dim3(NBOX), dim3(256), 0, stream,
                       x0, x1, x2, x3, boxes, bids, out);
}
